// ContinuousRNN_84430467105127
// MI455X (gfx1250) — matrix operand order repaired and checked by reading, not compiled, not run on hardware
//
#include <hip/hip_runtime.h>

#define NH    100
#define NPAD  112
#define TLEN  2048
#define BATCH 512
#define BT    16
#define DT_C  0.15f
#define STD_C 0.015f

#define NZ_SLOTS   4                    // LDS ring slots for noise tiles
#define NZ_DIST    3                    // prefetch distance (steps ahead)
#define NZ_SLOTB   (BT * NH * 4)        // bytes per noise tile (6400)
#define NZ_CHUNKS  (BT * NH / 4)        // 400 b128 chunks per tile

typedef __attribute__((ext_vector_type(16))) __bf16 v16bf;
typedef __attribute__((ext_vector_type(8)))  __bf16 bf16x8;
typedef __attribute__((ext_vector_type(8)))  float  v8f;

__device__ __forceinline__ unsigned short f2bf(float f) {
    union { float f; unsigned u; } v; v.f = f;
    unsigned r = v.u + 0x7FFFu + ((v.u >> 16) & 1u);   // round-to-nearest-even
    return (unsigned short)(r >> 16);
}

__device__ __forceinline__ float fast_tanh(float x) {
#if __has_builtin(__builtin_amdgcn_tanhf)
    return __builtin_amdgcn_tanhf(x);
#elif __has_builtin(__builtin_amdgcn_tanh_f32)
    return __builtin_amdgcn_tanh_f32(x);
#else
    float y;
    asm volatile("v_tanh_f32 %0, %1" : "=v"(y) : "v"(x));
    return y;
#endif
}

// ASYNCcnt-tracked global -> LDS copy (CDNA5 async path, §15.18 opcode 98).
// %0 = 32-bit LDS byte address (VDST), %1 = 64-bit global address (VADDR).
__device__ __forceinline__ void async_ld_b128(unsigned lds, const float* g) {
    asm volatile("global_load_async_to_lds_b128 %0, %1, off"
                 :: "v"(lds), "v"((unsigned long long)(size_t)g)
                 : "memory");
}

union BFV { v16bf v; unsigned short s[16]; };

__global__ __launch_bounds__(224) void rnn_scan_kernel(
    const float* __restrict__ inp,    // [B,T,3]
    const float* __restrict__ noise,  // [B,T,100]
    const float* __restrict__ wr,     // [100,100]  recurrent
    const float* __restrict__ wi,     // [100,3]    input weights
    const float* __restrict__ wo,     // [3,100]    output weights
    float* __restrict__ out)          // [B,T,3]
{
    // bf16 copy of h for WMMA A operand; stride 136 (272B) -> 16B aligned, bank-staggered
    __shared__ __align__(16) unsigned short hbf[BT][136];
    __shared__ float h32[BT][NPAD];                  // fp32 master state
    __shared__ float inT[BT][4];                     // inputs for current t
    __shared__ float woutS[3][NPAD];                 // output weights, zero padded
    __shared__ __align__(16) float noiseS[NZ_SLOTS][BT * NH];  // async noise ring

    const int tid  = threadIdx.x;
    const int wave = tid >> 5;         // 0..6 -> N tile
    const int lane = tid & 31;
    const int lo   = lane & 15;
    const int hi   = lane >> 4;
    const int bbase = blockIdx.x * BT;

    // ---------------- one-time init ----------------
    for (int i = tid; i < BT * 136;  i += 224) ((unsigned short*)hbf)[i] = 0;
    for (int i = tid; i < BT * NPAD; i += 224) ((float*)h32)[i] = 0.0f;
    for (int i = tid; i < 3 * NPAD;  i += 224) {
        int j = i / NPAD, n = i - j * NPAD;
        woutS[j][n] = (n < NH) ? wo[j * NH + n] : 0.0f;
    }
    if (tid < BT * 3) {   // stage inputs for t=0
        int r = tid / 3, k = tid - r * 3;
        inT[r][k] = inp[(size_t)(bbase + r) * TLEN * 3 + k];
    }

    // --- per-lane async-noise chunk assignment: chunks 2*tid and 2*tid+1 of 400 ---
    // (every wave, including wave 6, has active lanes in BOTH issues, so each wave
    //  issues exactly 2 ASYNCcnt instructions per prefetched step)
    const int  cid0 = 2 * tid, cid1 = 2 * tid + 1;
    const bool cv0  = cid0 < NZ_CHUNKS;
    const bool cv1  = cid1 < NZ_CHUNKS;
    const int  cr0  = cv0 ? cid0 / 25 : 0, cq0 = cv0 ? cid0 % 25 : 0;
    const int  cr1  = cv1 ? cid1 / 25 : 0, cq1 = cv1 ? cid1 % 25 : 0;
    const float*   gnz0 = noise + (size_t)(bbase + cr0) * TLEN * NH + cq0 * 4;
    const float*   gnz1 = noise + (size_t)(bbase + cr1) * TLEN * NH + cq1 * 4;
    const unsigned lnz0 = (unsigned)(size_t)&noiseS[0][cr0 * NH + cq0 * 4];
    const unsigned lnz1 = (unsigned)(size_t)&noiseS[0][cr1 * NH + cq1 * 4];

    const int ncol = wave * 16 + lo;          // this lane's output column n
    float wik0 = 0.f, wik1 = 0.f, wik2 = 0.f; // persistent input-weight row
    if (ncol < NH) {
        wik0 = wi[ncol * 3 + 0];
        wik1 = wi[ncol * 3 + 1];
        wik2 = wi[ncol * 3 + 2];
    }

    // Persistent B fragments: B[k][n] = W_rec[n][k]; ISA 16-bit B layout:
    // lanes 0-15 hold K = kt*32+0..7,16..23 (col = lane), lanes 16-31 hold K = kt*32+8..15,24..31.
    v16bf Bf[4];
    for (int kt = 0; kt < 4; ++kt) {
        BFV t;
        int kb = kt * 32 + (hi ? 8 : 0);
        #pragma unroll
        for (int j = 0; j < 16; ++j) {
            int k = kb + (j < 8 ? j : j + 8);
            float w = (ncol < NH && k < NH) ? wr[ncol * NH + k] : 0.0f;
            t.s[j] = f2bf(w);
        }
        Bf[kt] = t.v;
    }

    // ISA 16-bit A layout: lane row = lane%16; lanes 0-15: K base 0 (+0..7, +16..23),
    // lanes 16-31: K base 8.
    const int arow = lo;
    const int akb  = hi ? 8 : 0;

    // ---- prime the async noise pipeline: slots/steps 0..NZ_DIST-1 ----
    #pragma unroll
    for (int pt = 0; pt < NZ_DIST; ++pt) {
        if (cv0) async_ld_b128(lnz0 + (pt & (NZ_SLOTS - 1)) * NZ_SLOTB, gnz0 + (size_t)pt * NH);
        if (cv1) async_ld_b128(lnz1 + (pt & (NZ_SLOTS - 1)) * NZ_SLOTB, gnz1 + (size_t)pt * NH);
    }

    // ---------------- time scan ----------------
    for (int t = 0; t < TLEN; ++t) {
        __syncthreads();   // hbf/h32 from previous step + inT[t] visible; slot (t-1)%4 free

        // issue async noise prefetch for step t+NZ_DIST into its ring slot
        if (t + NZ_DIST < TLEN) {
            int slot = (t + NZ_DIST) & (NZ_SLOTS - 1);
            if (cv0) async_ld_b128(lnz0 + slot * NZ_SLOTB, gnz0 + (size_t)(t + NZ_DIST) * NH);
            if (cv1) async_ld_b128(lnz1 + slot * NZ_SLOTB, gnz1 + (size_t)(t + NZ_DIST) * NH);
        }

        // y = h @ W_rec^T for this wave's 16-column slice, K = 128 (zero padded)
        v8f acc = {0.f, 0.f, 0.f, 0.f, 0.f, 0.f, 0.f, 0.f};
        #pragma unroll
        for (int kt = 0; kt < 4; ++kt) {
            const unsigned short* p = &hbf[arow][kt * 32 + akb];
            bf16x8 alo = *reinterpret_cast<const bf16x8*>(p);        // K base..base+7
            bf16x8 ahi = *reinterpret_cast<const bf16x8*>(p + 16);   // K base+16..base+23
            v16bf A = __builtin_shufflevector(alo, ahi,
                0, 1, 2, 3, 4, 5, 6, 7, 8, 9, 10, 11, 12, 13, 14, 15);
            acc = __builtin_amdgcn_wmma_f32_16x16x32_bf16(
                false, A, false, Bf[kt], (short)0, acc, false, false);
        }

        // Async loads complete in order: <= 2*NZ_DIST outstanding ==> slot-t tile landed.
        if (t + NZ_DIST < TLEN) {
            asm volatile("s_wait_asynccnt 0x6" ::: "memory");
        } else {
            asm volatile("s_wait_asynccnt 0x0" ::: "memory");
        }

        __syncthreads();   // all waves: hbf reads done + slot-t noise visible

        // Euler update for this lane's 8 (m, ncol) elements (WMMA C layout: m = j + 8*hi)
        if (ncol < NH) {
            const float* nzrow = &noiseS[t & (NZ_SLOTS - 1)][0];
            #pragma unroll
            for (int j = 0; j < 8; ++j) {
                int m = j + hi * 8;
                float ip = inT[m][0] * wik0 + inT[m][1] * wik1 + inT[m][2] * wik2;
                float hprev = h32[m][ncol];
                float a  = fast_tanh(acc[j] + ip);
                float nz = nzrow[m * NH + ncol];
                float hnew = hprev + (a - hprev) * DT_C + nz * STD_C;
                h32[m][ncol] = hnew;
                hbf[m][ncol] = f2bf(hnew);
            }
        }

        __syncthreads();   // h32 updated for output projection

        if (tid < 48) {
            // out[b, t, j] = h[b, :] . wo[j, :]   (48 dots of length 112, pad = 0)
            int r = tid / 3, j = tid - r * 3;
            float s = 0.f;
            #pragma unroll 4
            for (int n = 0; n < NPAD; ++n) s += h32[r][n] * woutS[j][n];
            out[(size_t)(bbase + r) * TLEN * 3 + (size_t)t * 3 + j] = s;
        } else if (tid >= 64 && tid < 112 && (t + 1) < TLEN) {
            // prefetch inputs for t+1
            int idx = tid - 64;
            int r = idx / 3, k = idx - r * 3;
            inT[r][k] = inp[(size_t)(bbase + r) * TLEN * 3 + (size_t)(t + 1) * 3 + k];
        }
    }
}

extern "C" void kernel_launch(void* const* d_in, const int* in_sizes, int n_in,
                              void* d_out, int out_size, void* d_ws, size_t ws_size,
                              hipStream_t stream) {
    const float* inp   = (const float*)d_in[0];
    const float* noise = (const float*)d_in[1];
    const float* wr    = (const float*)d_in[2];
    const float* wi    = (const float*)d_in[3];
    const float* wo    = (const float*)d_in[4];
    float* out = (float*)d_out;

    dim3 grid(BATCH / BT);   // 32 blocks, each owns 16 batch rows for the full scan
    dim3 block(224);         // 7 waves: one 16-wide N tile per wave
    hipLaunchKernelGGL(rnn_scan_kernel, grid, block, 0, stream,
                       inp, noise, wr, wi, wo, out);
}
